// MultilayerGRU_4578435137548
// MI455X (gfx1250) — matrix operand order repaired and checked by reading, not compiled, not run on hardware
//
#include <hip/hip_runtime.h>

#define DEVINL __device__ __forceinline__

typedef __attribute__((ext_vector_type(16))) __bf16 bvec16;
typedef __attribute__((ext_vector_type(8)))  float  v8f;

union BV { uint4 u[2]; bvec16 v; };

DEVINL unsigned short f2bf(float f) {
  unsigned u = __builtin_bit_cast(unsigned, f);
  unsigned r = u + 0x7FFFu + ((u >> 16) & 1u);   // round-to-nearest-even
  return (unsigned short)(r >> 16);
}

DEVINL v8f wmma_bf16(BV a, BV b, v8f c) {
  // (neg_a, A, neg_b, B, c_mod, C, reuse_a, reuse_b)
  return __builtin_amdgcn_wmma_f32_16x16x32_bf16(false, a.v, false, b.v,
                                                 (short)0, c, false, false);
}

// A-operand (16x32 bf16) load from row-major [rows x ld] bf16 matrix.
// ISA layout: lane l: row = l&15; VGPR0-3 = K kb..kb+7, VGPR4-7 = K kb+16..kb+23,
// kb = (l>>4)*8.
DEVINL void loadA(BV& a, const unsigned short* __restrict__ src,
                  int row0, int ld, int k0, int lane) {
  int row = row0 + (lane & 15);
  int kb  = k0 + ((lane >> 4) << 3);
  const uint4* p = (const uint4*)(src + (size_t)row * ld + kb);
  a.u[0] = p[0];
  a.u[1] = p[2];  // +32 bytes = +16 bf16 elements
}

DEVINL float sigm(float x) { return 1.f / (1.f + __expf(-x)); }

DEVINL void gbar(unsigned* ctr, unsigned tgt) {
  __threadfence();
  __syncthreads();
  if (threadIdx.x == 0) {
    __atomic_fetch_add(ctr, 1u, __ATOMIC_RELEASE);
    while (__atomic_load_n(ctr, __ATOMIC_ACQUIRE) < tgt)
      __builtin_amdgcn_s_sleep(2);
  }
  __syncthreads();
  __threadfence();
}

// ---------------------------------------------------------------------------
// Pack f32 weight matrix W[K][N] (row-major) into WMMA B-operand tiles, bf16.
// Blob (nt*KT + kt): 1024 bytes; per lane 32 bytes at lane*32.
// Lane layout: n = nt*16 + (lane&15); dword i packs K = kb+2(i&3)+16(i>>2), +1 with
// kb = kt*32 + (lane>>4)*8.
// ---------------------------------------------------------------------------
__global__ void pack_wb_kernel(const float* __restrict__ W, uint4* __restrict__ dst,
                               int K, int N) {
  int t = blockIdx.x * blockDim.x + threadIdx.x;
  int KT = K >> 5, NT = N >> 4;
  int tile = t >> 5;
  if (tile >= KT * NT) return;
  int lane = t & 31;
  int kt = tile % KT, nt = tile / KT;
  int n  = (nt << 4) + (lane & 15);
  int kb = (kt << 5) + ((lane >> 4) << 3);
  unsigned d[8];
#pragma unroll
  for (int i = 0; i < 8; ++i) {
    int k0 = kb + 2 * (i & 3) + ((i >> 2) << 4);
    unsigned lo = f2bf(W[(size_t)k0 * N + n]);
    unsigned hi = f2bf(W[(size_t)(k0 + 1) * N + n]);
    d[i] = lo | (hi << 16);
  }
  uint4* o = dst + ((size_t)tile << 6) + (size_t)lane * 2;
  o[0] = make_uint4(d[0], d[1], d[2], d[3]);
  o[1] = make_uint4(d[4], d[5], d[6], d[7]);
}

__global__ void f32_to_bf16_kernel(const float* __restrict__ src,
                                   unsigned short* __restrict__ dst, size_t n) {
  size_t i = (size_t)blockIdx.x * blockDim.x + threadIdx.x;
  size_t stride = (size_t)gridDim.x * blockDim.x;
  for (; i < n; i += stride) dst[i] = f2bf(src[i]);
}

__global__ void init_h_kernel(const float* __restrict__ hs, float* __restrict__ h32,
                              unsigned short* __restrict__ hbf,
                              int layer, int L, int H, int B) {
  int i = blockIdx.x * blockDim.x + threadIdx.x;
  if (i >= B * H) return;
  int b = i / H, n = i % H;
  float v = hs[((size_t)b * L + layer) * H + n];
  h32[i] = v;
  hbf[i] = f2bf(v);
}

// ---------------------------------------------------------------------------
// GEMM: C[M][N] (f32) = A[M][K] (bf16, row-major) x Bpacked (+ optional bias).
// Block = 4 waves; wave computes 16 rows x 64 cols; grid (M/64, N/64).
// ---------------------------------------------------------------------------
__global__ void __launch_bounds__(128) wmma_gemm_bf16_kernel(
    const unsigned short* __restrict__ A, const uint4* __restrict__ Bp,
    const float* __restrict__ bias, float* __restrict__ C,
    int M, int N, int K) {
  const int lane = threadIdx.x & 31;
  const int wave = threadIdx.x >> 5;
  const int KT = K >> 5;
  const int m0 = blockIdx.x * 64 + wave * 16;
  const int nt0 = blockIdx.y * 4;
  v8f acc[4] = {};
  for (int kt = 0; kt < KT; ++kt) {
    BV a;
    loadA(a, A, m0, K, kt << 5, lane);
#pragma unroll
    for (int j = 0; j < 4; ++j) {
      const uint4* pb = Bp + ((size_t)(nt0 + j) * KT + kt) * 64 + (size_t)lane * 2;
      BV b; b.u[0] = pb[0]; b.u[1] = pb[1];
      acc[j] = wmma_bf16(a, b, acc[j]);
    }
    if (kt + 1 < KT)
      __builtin_prefetch(Bp + ((size_t)nt0 * KT + kt + 1) * 64 + (size_t)lane * 2, 0, 0);
  }
#pragma unroll
  for (int j = 0; j < 4; ++j) {
    int n = (nt0 + j) * 16 + (lane & 15);
    float bb = bias ? bias[n] : 0.f;
#pragma unroll
    for (int i = 0; i < 8; ++i) {
      int m = m0 + i + ((lane >> 4) << 3);
      C[(size_t)m * N + n] = acc[j][i] + bb;
    }
  }
}

// ---------------------------------------------------------------------------
// Persistent GRU scan. Grid = H/16 = 64 WGs, 128 threads (4 waves = 4 M-tiles
// of the B=64 batch). WG owns 16 hidden columns; weight slices (3 gates x
// H x 16 bf16 = 96KB) staged once in LDS. Two grid barriers per timestep.
// ---------------------------------------------------------------------------
__global__ void __launch_bounds__(128) gru_scan_kernel(
    const float* __restrict__ xz, const float* __restrict__ xr,
    const float* __restrict__ xg,
    const float* __restrict__ bz, const float* __restrict__ br,
    const float* __restrict__ bg,
    const uint4* __restrict__ wzp, const uint4* __restrict__ wrp,
    const uint4* __restrict__ wgp,
    float* __restrict__ h32, unsigned short* __restrict__ hbf,
    unsigned short* __restrict__ rhbf, unsigned short* __restrict__ ybf,
    float* __restrict__ hfin, unsigned* __restrict__ ctr,
    int Sdim, int Hdim, int LH) {
  extern __shared__ uint4 sW[];
  const int KT = Hdim >> 5;                    // 32
  const int lane = threadIdx.x & 31;
  const int wave = threadIdx.x >> 5;
  const int nt = blockIdx.x;
  const int n0 = nt << 4;
  const int mrow0 = wave << 4;
  const int cnt = KT * 64;                     // uint4 per gate slice

  {  // stage our column-slice of Whz/Whr/Whg into LDS (reused for all steps)
    const size_t base = (size_t)nt * cnt;
    for (int i = threadIdx.x; i < cnt; i += blockDim.x) {
      sW[i]           = wzp[base + i];
      sW[cnt + i]     = wrp[base + i];
      sW[2 * cnt + i] = wgp[base + i];
    }
  }
  __syncthreads();

  const int n = n0 + (lane & 15);
  const float bzs = bz[n], brs = br[n], bgs = bg[n];
  unsigned tgt = 0;
  const unsigned nwg = gridDim.x;

  for (int t = 0; t < Sdim; ++t) {
    // ---- phase A: z and r gates (h @ Whz, h @ Whr) ----
    v8f accZ = {}, accR = {};
    for (int kt = 0; kt < KT; ++kt) {
      BV a, bzt, brt;
      loadA(a, hbf, mrow0, Hdim, kt << 5, lane);
      const uint4* pz = sW + (size_t)kt * 64 + lane * 2;
      const uint4* pr = sW + cnt + (size_t)kt * 64 + lane * 2;
      bzt.u[0] = pz[0]; bzt.u[1] = pz[1];
      brt.u[0] = pr[0]; brt.u[1] = pr[1];
      accZ = wmma_bf16(a, bzt, accZ);
      accR = wmma_bf16(a, brt, accR);
    }
    float zs[8], hv[8];
#pragma unroll
    for (int i = 0; i < 8; ++i) {
      int b = mrow0 + i + ((lane >> 4) << 3);
      size_t xi = ((size_t)b * Sdim + t) * Hdim + n;
      float z = sigm(accZ[i] + xz[xi] + bzs);
      float r = sigm(accR[i] + xr[xi] + brs);
      float h = h32[(size_t)b * Hdim + n];
      zs[i] = z; hv[i] = h;
      rhbf[(size_t)b * Hdim + n] = f2bf(r * h);
    }
    tgt += nwg; gbar(ctr, tgt);

    // ---- phase B: g gate ((r*h) @ Whg) and state update ----
    v8f accG = {};
    for (int kt = 0; kt < KT; ++kt) {
      BV a, bgt;
      loadA(a, rhbf, mrow0, Hdim, kt << 5, lane);
      const uint4* pg = sW + 2 * cnt + (size_t)kt * 64 + lane * 2;
      bgt.u[0] = pg[0]; bgt.u[1] = pg[1];
      accG = wmma_bf16(a, bgt, accG);
    }
#pragma unroll
    for (int i = 0; i < 8; ++i) {
      int b = mrow0 + i + ((lane >> 4) << 3);
      size_t xi = ((size_t)b * Sdim + t) * Hdim + n;
      float g = tanhf(accG[i] + xg[xi] + bgs);
      float hn = zs[i] * hv[i] + (1.f - zs[i]) * g;
      h32[(size_t)b * Hdim + n] = hn;
      hbf[(size_t)b * Hdim + n] = f2bf(hn);
      ybf[xi] = f2bf(hn);
      if (t == Sdim - 1) hfin[(size_t)b * LH + n] = hn;
    }
    tgt += nwg; gbar(ctr, tgt);
  }
}

// ---------------------------------------------------------------------------
extern "C" void kernel_launch(void* const* d_in, const int* in_sizes, int n_in,
                              void* d_out, int out_size, void* d_ws, size_t ws_size,
                              hipStream_t stream) {
  (void)in_sizes; (void)n_in; (void)out_size; (void)ws_size;
  constexpr int B = 64, S = 512, I = 512, H = 1024, O = 512, L = 2;
  constexpr size_t BS = (size_t)B * S;

  const float* x   = (const float*)d_in[0];
  const float* hs  = (const float*)d_in[1];
  const float* wxz[2] = { (const float*)d_in[2],  (const float*)d_in[11] };
  const float* whz[2] = { (const float*)d_in[3],  (const float*)d_in[12] };
  const float* bhz[2] = { (const float*)d_in[4],  (const float*)d_in[13] };
  const float* wxr[2] = { (const float*)d_in[5],  (const float*)d_in[14] };
  const float* whr[2] = { (const float*)d_in[6],  (const float*)d_in[15] };
  const float* bhr[2] = { (const float*)d_in[7],  (const float*)d_in[16] };
  const float* wxg[2] = { (const float*)d_in[8],  (const float*)d_in[17] };
  const float* whg[2] = { (const float*)d_in[9],  (const float*)d_in[18] };
  const float* bhg[2] = { (const float*)d_in[10], (const float*)d_in[19] };
  const float* why_w = (const float*)d_in[20];
  const float* why_b = (const float*)d_in[21];
  float* out = (float*)d_out;

  // ---- workspace carve-up ----
  char* ws = (char*)d_ws;
  size_t off = 0;
  auto take = [&](size_t bytes) -> char* {
    char* p = ws + off;
    off = (off + bytes + 255) & ~(size_t)255;
    return p;
  };
  float* xzb = (float*)take(BS * H * 4);
  float* xrb = (float*)take(BS * H * 4);
  float* xgb = (float*)take(BS * H * 4);
  unsigned short* xbf  = (unsigned short*)take(BS * I * 2);
  unsigned short* y0bf = (unsigned short*)take(BS * H * 2);
  unsigned short* y1bf = (unsigned short*)take(BS * H * 2);
  float* h32           = (float*)take((size_t)B * H * 4);
  unsigned short* hbf  = (unsigned short*)take((size_t)B * H * 2);
  unsigned short* rhbf = (unsigned short*)take((size_t)B * H * 2);
  auto packed_bytes = [](int K, int N) { return (size_t)(K / 32) * (N / 16) * 1024; };
  uint4* wxzp[2], *wxrp[2], *wxgp[2], *whzp[2], *whrp[2], *whgp[2];
  for (int l = 0; l < 2; ++l) {
    int Kx = (l == 0) ? I : H;
    wxzp[l] = (uint4*)take(packed_bytes(Kx, H));
    wxrp[l] = (uint4*)take(packed_bytes(Kx, H));
    wxgp[l] = (uint4*)take(packed_bytes(Kx, H));
    whzp[l] = (uint4*)take(packed_bytes(H, H));
    whrp[l] = (uint4*)take(packed_bytes(H, H));
    whgp[l] = (uint4*)take(packed_bytes(H, H));
  }
  uint4* whyp = (uint4*)take(packed_bytes(H, O));
  unsigned* ctr = (unsigned*)take(256);

  // ---- pack all weights into WMMA B-operand layout (bf16) ----
  auto pack = [&](const float* W, uint4* dst, int K, int N) {
    int threads = (K >> 5) * (N >> 4) * 32;
    pack_wb_kernel<<<dim3((threads + 127) / 128), dim3(128), 0, stream>>>(W, dst, K, N);
  };
  for (int l = 0; l < 2; ++l) {
    int Kx = (l == 0) ? I : H;
    pack(wxz[l], wxzp[l], Kx, H);
    pack(wxr[l], wxrp[l], Kx, H);
    pack(wxg[l], wxgp[l], Kx, H);
    pack(whz[l], whzp[l], H, H);
    pack(whr[l], whrp[l], H, H);
    pack(whg[l], whgp[l], H, H);
  }
  pack(why_w, whyp, H, O);

  // ---- bf16 copy of the input ----
  f32_to_bf16_kernel<<<dim3(2048), dim3(256), 0, stream>>>(x, xbf, BS * I);

  auto gemm = [&](const unsigned short* A, const uint4* Bp, const float* bias,
                  float* C, int M, int N, int K) {
    wmma_gemm_bf16_kernel<<<dim3(M / 64, N / 64), dim3(128), 0, stream>>>(
        A, Bp, bias, C, M, N, K);
  };

  const unsigned short* lin[2] = { xbf, y0bf };
  unsigned short* lout[2] = { y0bf, y1bf };
  float* hfin_base = out + BS * O;  // hidden_out (B, L, H)

  for (int l = 0; l < 2; ++l) {
    int Kx = (l == 0) ? I : H;
    // x-projections for all timesteps (parallel GEMMs)
    gemm(lin[l], wxzp[l], nullptr, xzb, (int)BS, H, Kx);
    gemm(lin[l], wxrp[l], nullptr, xrb, (int)BS, H, Kx);
    gemm(lin[l], wxgp[l], nullptr, xgb, (int)BS, H, Kx);
    // recurrent state init
    init_h_kernel<<<dim3((B * H + 127) / 128), dim3(128), 0, stream>>>(
        hs, h32, hbf, l, L, H, B);
    // persistent sequential scan
    hipMemsetAsync(ctr, 0, sizeof(unsigned), stream);
    gru_scan_kernel<<<dim3(H / 16), dim3(128), 3 * (H / 32) * 1024, stream>>>(
        xzb, xrb, xgb, bhz[l], bhr[l], bhg[l],
        whzp[l], whrp[l], whgp[l],
        h32, hbf, rhbf, lout[l], hfin_base + (size_t)l * H, ctr,
        S, H, L * H);
  }

  // output projection with bias
  gemm(y1bf, whyp, why_b, out, (int)BS, O, H);
}
